// PrefillAttention_40054865002723
// MI455X (gfx1250) — matrix operand order repaired and checked by reading, not compiled, not run on hardware
//
#include <hip/hip_runtime.h>

typedef __attribute__((ext_vector_type(16))) _Float16 v16h;
typedef __attribute__((ext_vector_type(8)))  float    v8f;
typedef __attribute__((ext_vector_type(4)))  int      v4i;

constexpr int Bdim = 2, Sdim = 2048, Edim = 1024, Hdim = 16, Ddim = 64;
constexpr int Mrows = Bdim * Sdim;                       // 4096
constexpr size_t QH = (size_t)Bdim * Hdim * Sdim * Ddim; // 4194304 elems / tensor

union AF { unsigned u[8]; v16h v; };
union CF { float f[8]; v8f v; };

__device__ inline unsigned short f2h(float x) {
    _Float16 h = (_Float16)x;
    return __builtin_bit_cast(unsigned short, h);
}
__device__ inline unsigned pack2(float a, float b) {
    return (unsigned)f2h(a) | ((unsigned)f2h(b) << 16);
}

// --- CDNA5 async global->LDS copy (ASYNCcnt path), guarded ------------------
#if defined(__has_builtin)
#if __has_builtin(__builtin_amdgcn_global_load_async_to_lds_b128) && \
    __has_builtin(__builtin_amdgcn_s_wait_asynccnt)
#define USE_ASYNC 1
#endif
#endif
#ifndef USE_ASYNC
#define USE_ASYNC 0
#endif

typedef __attribute__((address_space(1))) v4i gv4i;
typedef __attribute__((address_space(3))) v4i lv4i;

__device__ inline void async_cp16(const _Float16* g, _Float16* l) {
#if USE_ASYNC
    __builtin_amdgcn_global_load_async_to_lds_b128(
        (gv4i*)g, (lv4i*)l, 0, 0);
#endif
}

// ---------------------------------------------------------------------------
// GEMM: C[M x N] = A[M x 1024] * W[1024 x N] + bias
// MODE 0: N=3072 (QKV). Epilogue: K,V fp32 -> d_out caches; fp16 -> ws:
//         Q,K in [B,H,S,D]; V transposed in [B,H,D,S].
// MODE 1: N=1024 (proj). Epilogue: fp32 -> d_out.
// Block 256 thr (8 waves as 2Mx4N). Tile 64(M) x 256(N), BK=32.
// Wave: 32x64 strip -> 2 A-frags x 4 B-frags -> 8 WMMAs / K-step.
// ---------------------------------------------------------------------------
template <int MODE>
__global__ __launch_bounds__(256) void gemm_kernel(
    const float* __restrict__ A, const float* __restrict__ W,
    const float* __restrict__ bias, float* __restrict__ dout,
    _Float16* __restrict__ wsH)
{
    constexpr int NT = (MODE == 0) ? 3 * Edim : Edim;
    __shared__ _Float16 As[64][34];    // [m][k], +2 pad (17-dword rows)
    __shared__ _Float16 Bs[256][34];   // [n][k] transposed, +2 pad

    const int tid = threadIdx.x;
    const int lane = tid & 31, w = tid >> 5;
    const int m0 = (w & 1) * 32, n0 = (w >> 1) * 64;
    const int gmBase = blockIdx.y * 64, gnBase = blockIdx.x * 256;

    CF acc[2][4];
#pragma unroll
    for (int i = 0; i < 2; i++)
#pragma unroll
        for (int j = 0; j < 4; j++)
#pragma unroll
            for (int r = 0; r < 8; r++) acc[i][j].f[r] = 0.0f;

    for (int kk = 0; kk < Edim; kk += 32) {
        // A tile: 64 x 32 fp32 -> fp16
#pragma unroll
        for (int p = 0; p < 2; p++) {
            const int idx = tid + p * 256;
            const int r = idx >> 3, c = (idx & 7) * 4;
            const float4 f = *(const float4*)(A + (size_t)(gmBase + r) * Edim + kk + c);
            *(unsigned*)&As[r][c]     = pack2(f.x, f.y);
            *(unsigned*)&As[r][c + 2] = pack2(f.z, f.w);
        }
        // B tile: 32 x 256 fp32, stored transposed [n][k]
#pragma unroll
        for (int p = 0; p < 8; p++) {
            const int idx = tid + p * 256;
            const int r = idx >> 6, cb = (idx & 63) * 4;
            const float4 f = *(const float4*)(W + (size_t)(kk + r) * NT + gnBase + cb);
            Bs[cb + 0][r] = (_Float16)f.x;
            Bs[cb + 1][r] = (_Float16)f.y;
            Bs[cb + 2][r] = (_Float16)f.z;
            Bs[cb + 3][r] = (_Float16)f.w;
        }
        __syncthreads();

        AF a[2];
#pragma unroll
        for (int mf = 0; mf < 2; mf++) {
            const int m  = m0 + mf * 16 + (lane & 15);
            const int kb = (lane >> 4) * 8;
#pragma unroll
            for (int j = 0; j < 8; j++) {
                const int k = ((j >> 2) << 4) + ((j & 3) << 1) + kb;
                a[mf].u[j] = *(const unsigned*)&As[m][k];
            }
        }
#pragma unroll
        for (int nf = 0; nf < 4; nf++) {
            AF bfr;
            const int n  = n0 + nf * 16 + (lane & 15);
            const int kb = (lane >> 4) * 8;
#pragma unroll
            for (int j = 0; j < 8; j++)
                bfr.u[j] = *(const unsigned*)&Bs[n][kb + ((j >> 2) << 4) + ((j & 3) << 1)];
#pragma unroll
            for (int mf = 0; mf < 2; mf++)
                acc[mf][nf].v = __builtin_amdgcn_wmma_f32_16x16x32_f16(
                    false, a[mf].v, false, bfr.v, (short)0, acc[mf][nf].v,
                    false, false);
        }
        __syncthreads();
    }

    // Epilogue
#pragma unroll
    for (int nf = 0; nf < 4; nf++) {
        const int gn = gnBase + n0 + nf * 16 + (lane & 15);
        const float bv = bias[gn];
#pragma unroll
        for (int mf = 0; mf < 2; mf++) {
#pragma unroll
            for (int r = 0; r < 8; r++) {
                const int gm = gmBase + m0 + mf * 16 + r + (lane >> 4) * 8;
                const float val = acc[mf][nf].f[r] + bv;
                if (MODE == 0) {
                    const int which = gn >> 10;       // 0:Q 1:K 2:V
                    const int e  = gn & 1023;
                    const int hh = e >> 6, dd = e & 63;
                    const int bb = gm >> 11, ss = gm & 2047;
                    size_t hoff;
                    if (which == 2)   // V transposed: [B,H,D,S]
                        hoff = 2 * QH +
                            ((size_t)(bb * Hdim + hh) * Ddim + dd) * Sdim + ss;
                    else              // Q,K: [B,H,S,D]
                        hoff = (size_t)which * QH +
                            ((size_t)(bb * Hdim + hh) * Sdim + ss) * Ddim + dd;
                    wsH[hoff] = (_Float16)val;
                    if (which == 1)
                        dout[(size_t)Mrows * Edim + (size_t)gm * Edim + e] = val;
                    else if (which == 2)
                        dout[2 * (size_t)Mrows * Edim + (size_t)gm * Edim + e] = val;
                } else {
                    dout[(size_t)gm * Edim + gn] = val;
                }
            }
        }
    }
}

// ---------------------------------------------------------------------------
// Flash attention. Grid (S/128, H, B). Block 256 (8 waves, 16 q-rows each).
// 64-key tiles: Ks[key][d], Vs[d][key] (V pre-transposed in ws) loaded with
// async global->LDS b128 copies when available.
// ---------------------------------------------------------------------------
__global__ __launch_bounds__(256) void attn_kernel(
    const _Float16* __restrict__ qh, const _Float16* __restrict__ kh,
    const _Float16* __restrict__ vT, const float* __restrict__ abias,
    float* __restrict__ ctxBuf)
{
    __shared__ _Float16 Ks[64][72];    // 144B rows: b128-aligned, conflict-free
    __shared__ _Float16 Vs[64][72];
    __shared__ _Float16 Ps[8][16][66];

    const int tid = threadIdx.x, lane = tid & 31, w = tid >> 5;
    const int h = blockIdx.y, b = blockIdx.z;
    const int qblk = blockIdx.x * 128;
    const int qbase = qblk + w * 16;
    const size_t headOff = ((size_t)(b * Hdim + h)) * Sdim * Ddim;
    const _Float16* Qp  = qh + headOff;
    const _Float16* Kp  = kh + headOff;
    const _Float16* VpT = vT + headOff;   // [D][S] per head

    // Q fragments: 16x64 -> two 16x32 A-fragments in registers
    AF qf[2];
    {
        const int m  = lane & 15;
        const int kb = (lane >> 4) * 8;
        const _Float16* row = Qp + (size_t)(qbase + m) * Ddim;
#pragma unroll
        for (int f = 0; f < 2; f++)
#pragma unroll
            for (int j = 0; j < 8; j++) {
                const int k = f * 32 + ((j >> 2) << 4) + ((j & 3) << 1) + kb;
                qf[f].u[j] = *(const unsigned*)(row + k);
            }
    }

    CF ctx[4];
#pragma unroll
    for (int i = 0; i < 4; i++)
#pragma unroll
        for (int j = 0; j < 8; j++) ctx[i].f[j] = 0.0f;
    float mrow[8], lrow[8];
#pragma unroll
    for (int r = 0; r < 8; r++) { mrow[r] = -3.0e30f; lrow[r] = 0.0f; }

    const int kend = qblk + 128;   // causal upper bound for this block
    for (int kk = 0; kk < kend; kk += 64) {
        // --- K/V tile -> LDS ------------------------------------------------
#if USE_ASYNC
#pragma unroll
        for (int p = 0; p < 2; p++) {
            const int idx = tid + p * 256;
            const int r = idx >> 3, cq = idx & 7;      // r: row, cq: 8-half col
            async_cp16(Kp  + (size_t)(kk + r) * Ddim + cq * 8, &Ks[r][cq * 8]);
            async_cp16(VpT + (size_t)r * Sdim + kk + cq * 8,   &Vs[r][cq * 8]);
        }
        __builtin_amdgcn_s_wait_asynccnt(0);
#else
#pragma unroll
        for (int p = 0; p < 2; p++) {
            const int idx = tid + p * 256;
            const int r = idx >> 3, cq = idx & 7;
            union { uint4 q; unsigned u[4]; } kv, vv;
            kv.q = *(const uint4*)(Kp  + (size_t)(kk + r) * Ddim + cq * 8);
            vv.q = *(const uint4*)(VpT + (size_t)r * Sdim + kk + cq * 8);
#pragma unroll
            for (int i = 0; i < 4; i++) {
                *(unsigned*)&Ks[r][cq * 8 + 2 * i] = kv.u[i];
                *(unsigned*)&Vs[r][cq * 8 + 2 * i] = vv.u[i];
            }
        }
#endif
        __syncthreads();

        // --- scores: 4 key-fragments of 16, each = 2 WMMAs over D=64 --------
        float pv[4][8];
        float tmax[8];
#pragma unroll
        for (int r = 0; r < 8; r++) tmax[r] = -3.0e30f;
#pragma unroll
        for (int kf = 0; kf < 4; kf++) {
            CF sc;
#pragma unroll
            for (int j = 0; j < 8; j++) sc.f[j] = 0.0f;
#pragma unroll
            for (int f = 0; f < 2; f++) {
                AF bfr;
                const int n  = kf * 16 + (lane & 15);
                const int kb = f * 32 + (lane >> 4) * 8;
#pragma unroll
                for (int j = 0; j < 8; j++)
                    bfr.u[j] = *(const unsigned*)&Ks[n][kb + ((j >> 2) << 4) + ((j & 3) << 1)];
                sc.v = __builtin_amdgcn_wmma_f32_16x16x32_f16(
                    false, qf[f].v, false, bfr.v, (short)0, sc.v, false, false);
            }
            const int gk = kk + kf * 16 + (lane & 15);
#pragma unroll
            for (int r = 0; r < 8; r++) {
                const int gq = qbase + r + (lane >> 4) * 8;
                float v = sc.f[r] * 0.125f + abias[(size_t)gq * Sdim + gk];
                if (gk > gq) v = -1.0e9f;     // causal mask
                pv[kf][r] = v;
                tmax[r] = fmaxf(tmax[r], v);
            }
        }
        // row max across 16 lanes (xor masks stay inside each half)
#pragma unroll
        for (int r = 0; r < 8; r++) {
            float v = tmax[r];
            v = fmaxf(v, __shfl_xor(v, 1, 32));
            v = fmaxf(v, __shfl_xor(v, 2, 32));
            v = fmaxf(v, __shfl_xor(v, 4, 32));
            v = fmaxf(v, __shfl_xor(v, 8, 32));
            tmax[r] = v;
        }
        // online softmax update
        float rs[8];
#pragma unroll
        for (int r = 0; r < 8; r++) {
            const float mnew = fmaxf(mrow[r], tmax[r]);
            const float scl  = __expf(mrow[r] - mnew);
            mrow[r] = mnew;
            float sum = 0.0f;
#pragma unroll
            for (int kf = 0; kf < 4; kf++) {
                const float p = __expf(pv[kf][r] - mnew);
                pv[kf][r] = p;
                sum += p;
            }
            sum += __shfl_xor(sum, 1, 32);
            sum += __shfl_xor(sum, 2, 32);
            sum += __shfl_xor(sum, 4, 32);
            sum += __shfl_xor(sum, 8, 32);
            lrow[r] = lrow[r] * scl + sum;
            rs[r] = scl;
        }
        // rescale ctx accumulators (row mapping identical to score layout)
#pragma unroll
        for (int df = 0; df < 4; df++)
#pragma unroll
            for (int r = 0; r < 8; r++) ctx[df].f[r] *= rs[r];
        // stage P (C-layout -> wave-private LDS -> A-layout)
#pragma unroll
        for (int kf = 0; kf < 4; kf++)
#pragma unroll
            for (int r = 0; r < 8; r++)
                Ps[w][r + (lane >> 4) * 8][kf * 16 + (lane & 15)] =
                    (_Float16)pv[kf][r];
        // ctx += P(16x64) @ V(64x64): 2 k-steps x 4 d-fragments
#pragma unroll
        for (int ks = 0; ks < 2; ks++) {
            AF pf;
            {
                const int m  = lane & 15;
                const int kb = (lane >> 4) * 8;
#pragma unroll
                for (int j = 0; j < 8; j++) {
                    const int k = ks * 32 + ((j >> 2) << 4) + ((j & 3) << 1) + kb;
                    pf.u[j] = *(const unsigned*)&Ps[w][m][k];
                }
            }
#pragma unroll
            for (int df = 0; df < 4; df++) {
                AF vf;
                const int n  = df * 16 + (lane & 15);
                const int kb = ks * 32 + (lane >> 4) * 8;
#pragma unroll
                for (int j = 0; j < 8; j++)
                    vf.u[j] = *(const unsigned*)&Vs[n][kb + ((j >> 2) << 4) + ((j & 3) << 1)];
                ctx[df].v = __builtin_amdgcn_wmma_f32_16x16x32_f16(
                    false, pf.v, false, vf.v, (short)0, ctx[df].v, false, false);
            }
        }
        __syncthreads();
    }

    // finalize: ctx / l -> ws ctx buffer [B,S,E] with e = h*64 + d
#pragma unroll
    for (int r = 0; r < 8; r++) lrow[r] = 1.0f / lrow[r];
#pragma unroll
    for (int df = 0; df < 4; df++) {
        const int e = h * 64 + df * 16 + (lane & 15);
#pragma unroll
        for (int r = 0; r < 8; r++) {
            const int s = qbase + r + (lane >> 4) * 8;
            ctxBuf[((size_t)(b * Sdim + s)) * Edim + e] = ctx[df].f[r] * lrow[r];
        }
    }
}

// ---------------------------------------------------------------------------
extern "C" void kernel_launch(void* const* d_in, const int* in_sizes, int n_in,
                              void* d_out, int out_size, void* d_ws, size_t ws_size,
                              hipStream_t stream)
{
    const float* hidden = (const float*)d_in[0];
    const float* abias  = (const float*)d_in[1];
    const float* Wqkv   = (const float*)d_in[2];
    const float* bqkv   = (const float*)d_in[3];
    const float* Wproj  = (const float*)d_in[4];
    const float* bproj  = (const float*)d_in[5];
    float* dout = (float*)d_out;

    _Float16* wsH = (_Float16*)d_ws;           // q/k/vT fp16: 3 * QH elems (24 MB)
    float* ctxBuf = (float*)(wsH + 3 * QH);    // ctx fp32: 16 MB

    // 1) QKV GEMM + cache write + fp16 head-major staging (V transposed)
    gemm_kernel<0><<<dim3(3 * Edim / 256, Mrows / 64), 256, 0, stream>>>(
        hidden, Wqkv, bqkv, dout, wsH);
    // 2) flash attention -> ctx
    attn_kernel<<<dim3(Sdim / 128, Hdim, Bdim), 256, 0, stream>>>(
        wsH, wsH + QH, wsH + 2 * QH, abias, ctxBuf);
    // 3) output projection
    gemm_kernel<1><<<dim3(Edim / 256, Mrows / 64), 256, 0, stream>>>(
        ctxBuf, Wproj, bproj, dout, wsH);
}
